// learn_SLIC_calc_v1_new_54288386621759
// MI455X (gfx1250) — matrix operand order repaired and checked by reading, not compiled, not run on hardware
//
#include <hip/hip_runtime.h>

// Problem constants (match reference)
#define BSZ   4
#define MPTS  1024
#define NPTS  65536
#define KNB   6
#define CIN   64
#define COUT  32

typedef __attribute__((ext_vector_type(16))) __bf16 v16bf;
typedef __attribute__((ext_vector_type(8)))  float  v8f;

// ---- LDS partition (floats) ----
// accfea : 1024*64 = 65536
// accw   : 1024
// accxyz : 1024*3  = 3072
// cWx    : 96, cgx: 32, cbx: 32
// ystage : 8 waves * 16 rows * 33 (padded) = 4224
// pstage : 4224
#define OFF_ACCW   65536
#define OFF_ACCXYZ 66560
#define OFF_CWX    69632
#define OFF_CGX    69728
#define OFF_CBX    69760
#define OFF_YST    69792
#define OFF_PST    74016
#define SM_FLOATS  78240
#define YSTRIDE    33

__device__ __forceinline__ void wave_lds_fence() {
  __builtin_amdgcn_wave_barrier();
  asm volatile("s_wait_dscnt 0" ::: "memory");
  __builtin_amdgcn_wave_barrier();
}

__device__ __forceinline__ v8f wmma_bf16(v16bf a, v16bf b, v8f c) {
  // D = A(16x32 bf16) x B(32x16 bf16) + C(16x16 f32)
  return __builtin_amdgcn_wmma_f32_16x16x32_bf16(
      false, a, false, b, (short)0, c, false, false);
}

// A-fragment K index for VGPR-pair v (ISA 7.12.2, 16-bit A 16x32)
__device__ __forceinline__ int a_kbase(int t, int hf, int v) {
  return t * 32 + 8 * hf + ((v < 4) ? (2 * v) : (16 + 2 * (v - 4)));
}

__global__ void __launch_bounds__(256) slic_main(
    const float* __restrict__ sp_fea, const float* __restrict__ sp_xyz,
    const float* __restrict__ o_p_fea, const float* __restrict__ p_xyz,
    const int* __restrict__ c2p_idx_abs, const int* __restrict__ c2p_idx,
    const float* __restrict__ W_fea, const float* __restrict__ g_fea,
    const float* __restrict__ b_fea, const float* __restrict__ W_xyz,
    const float* __restrict__ g_xyz, const float* __restrict__ b_xyz,
    const float* __restrict__ W_mlp, const float* __restrict__ g_mlp,
    const float* __restrict__ b_mlp,
    float* __restrict__ ws_fea, float* __restrict__ ws_w,
    float* __restrict__ ws_xyz) {
  extern __shared__ float sm[];
  float* accfea = sm;
  float* accw   = sm + OFF_ACCW;
  float* accxyz = sm + OFF_ACCXYZ;
  float* cWx    = sm + OFF_CWX;
  float* cgx    = sm + OFF_CGX;
  float* cbx    = sm + OFF_CBX;

  const int tid  = threadIdx.x;
  const int lane = tid & 31;
  const int wv   = tid >> 5;
  const int b    = blockIdx.y;

  // zero the LDS segment accumulators, stage small xyz-MLP constants
  for (int i = tid; i < OFF_CWX; i += 256) sm[i] = 0.f;
  if (tid < 96) cWx[tid] = W_xyz[tid];
  if (tid < 32) { cgx[tid] = g_xyz[tid]; cbx[tid] = b_xyz[tid]; }
  __syncthreads();

  const int hf = lane >> 4;   // half-wave select
  const int ml = lane & 15;   // row (A) or column (B/D) index

  // --- preload B fragments (W_mlp, W_fea) : t = K-block, cb = N-block ---
  v16bf Bm[2][2], Bf[2][2];
#pragma unroll
  for (int t = 0; t < 2; ++t)
#pragma unroll
    for (int cb = 0; cb < 2; ++cb) {
      const int n  = cb * 16 + ml;
      const int kb = t * 32 + 8 * hf;
      v16bf vm, vf;
#pragma unroll
      for (int v = 0; v < 8; ++v) {
        const int ko = (v < 4) ? (2 * v) : (2 * v + 8);
        vm[2 * v]     = (__bf16)W_mlp[n * CIN + kb + ko];
        vm[2 * v + 1] = (__bf16)W_mlp[n * CIN + kb + ko + 1];
        vf[2 * v]     = (__bf16)W_fea[n * CIN + kb + ko];
        vf[2 * v + 1] = (__bf16)W_fea[n * CIN + kb + ko + 1];
      }
      Bm[t][cb] = vm;
      Bf[t][cb] = vf;
    }

  // per-lane BN scale/shift for WMMA output channel = ml + 16*cb
  float gmc[2], bmc[2], gfc[2], bfc[2];
#pragma unroll
  for (int cb = 0; cb < 2; ++cb) {
    gmc[cb] = g_mlp[ml + 16 * cb];
    bmc[cb] = b_mlp[ml + 16 * cb];
    gfc[cb] = g_fea[ml + 16 * cb];
    bfc[cb] = b_fea[ml + 16 * cb];
  }

  float* yst = sm + OFF_YST + wv * (16 * YSTRIDE);
  float* pst = sm + OFF_PST + wv * (16 * YSTRIDE);

  for (int it = 0; it < 16; ++it) {
    const int n0   = blockIdx.x * 2048 + it * 128 + wv * 16;
    const int prow = n0 + ml;  // this lane's point row
    const float* oprow = o_p_fea + (b * NPTS + prow) * CIN;

    // ---------- p_fea = l2norm(relu(o_p_fea @ W_mlp^T * g + b)) ----------
    v8f p0 = {}, p1 = {};
#pragma unroll
    for (int t = 0; t < 2; ++t) {
      v16bf a;
#pragma unroll
      for (int v = 0; v < 8; ++v) {
        const int kb  = a_kbase(t, hf, v);
        a[2 * v]     = (__bf16)oprow[kb];
        a[2 * v + 1] = (__bf16)oprow[kb + 1];
      }
      p0 = wmma_bf16(a, Bm[t][0], p0);
      p1 = wmma_bf16(a, Bm[t][1], p1);
    }
#pragma unroll
    for (int e = 0; e < 8; ++e) {
      const int r = e + 8 * hf;
      yst[r * YSTRIDE + ml]      = fmaxf(p0[e] * gmc[0] + bmc[0], 0.f);
      yst[r * YSTRIDE + ml + 16] = fmaxf(p1[e] * gmc[1] + bmc[1], 0.f);
    }
    wave_lds_fence();
    if (lane < 16) {
      float ss = 0.f;
#pragma unroll
      for (int c = 0; c < 32; ++c) {
        const float y = yst[lane * YSTRIDE + c];
        ss += y * y;
      }
      const float inv = 1.f / fmaxf(sqrtf(ss), 1e-12f);
#pragma unroll
      for (int c = 0; c < 32; ++c)
        pst[lane * YSTRIDE + c] = yst[lane * YSTRIDE + c] * inv;
    }
    wave_lds_fence();

    // ---------- per-neighbor bilinear weights ----------
    float wfl[6] = {0, 0, 0, 0, 0, 0};
    float wxl[6] = {0, 0, 0, 0, 0, 0};
    int   segl[6] = {0, 0, 0, 0, 0, 0};
#pragma unroll
    for (int j = 0; j < KNB; ++j) {
      const int gidx = c2p_idx_abs[(b * NPTS + prow) * KNB + j];
      const float* sprow = sp_fea + (b * MPTS + gidx) * CIN;
      v8f f0 = {}, f1 = {};
#pragma unroll
      for (int t = 0; t < 2; ++t) {
        v16bf a;
#pragma unroll
        for (int v = 0; v < 8; ++v) {
          const int kb  = a_kbase(t, hf, v);
          a[2 * v]     = (__bf16)(sprow[kb] - oprow[kb]);
          a[2 * v + 1] = (__bf16)(sprow[kb + 1] - oprow[kb + 1]);
        }
        f0 = wmma_bf16(a, Bf[t][0], f0);
        f1 = wmma_bf16(a, Bf[t][1], f1);
      }
#pragma unroll
      for (int e = 0; e < 8; ++e) {
        const int r = e + 8 * hf;
        yst[r * YSTRIDE + ml]      = fmaxf(f0[e] * gfc[0] + bfc[0], 0.f);
        yst[r * YSTRIDE + ml + 16] = fmaxf(f1[e] * gfc[1] + bfc[1], 0.f);
      }
      wave_lds_fence();
      if (lane < 16) {
        float ss = 0.f, dt = 0.f;
#pragma unroll
        for (int c = 0; c < 32; ++c) {
          const float y = yst[lane * YSTRIDE + c];
          ss += y * y;
          dt += y * pst[lane * YSTRIDE + c];
        }
        wfl[j] = dt / fmaxf(sqrtf(ss), 1e-12f);
        // tiny xyz branch (3 -> 32) on VALU
        float dx0 = sp_xyz[(b * MPTS + gidx) * 3 + 0] - p_xyz[(b * NPTS + prow) * 3 + 0];
        float dx1 = sp_xyz[(b * MPTS + gidx) * 3 + 1] - p_xyz[(b * NPTS + prow) * 3 + 1];
        float dx2 = sp_xyz[(b * MPTS + gidx) * 3 + 2] - p_xyz[(b * NPTS + prow) * 3 + 2];
        float ssx = 0.f, dtx = 0.f;
#pragma unroll
        for (int c = 0; c < 32; ++c) {
          float y = cWx[c * 3 + 0] * dx0 + cWx[c * 3 + 1] * dx1 + cWx[c * 3 + 2] * dx2;
          y = fmaxf(y * cgx[c] + cbx[c], 0.f);
          ssx += y * y;
          dtx += y * pst[lane * YSTRIDE + c];
        }
        wxl[j]  = dtx / fmaxf(sqrtf(ssx), 1e-12f);
        segl[j] = c2p_idx[(b * NPTS + prow) * KNB + j];
      }
      wave_lds_fence();
    }

    // ---------- softmax over 6 neighbors (lanes 0..15 own one row each) ----------
    float biw[6] = {0, 0, 0, 0, 0, 0};
    if (lane < 16) {
      float lg[6], mx = -3.4e38f;
#pragma unroll
      for (int j = 0; j < 6; ++j) { lg[j] = wfl[j] * wxl[j]; mx = fmaxf(mx, lg[j]); }
      float s = 0.f;
#pragma unroll
      for (int j = 0; j < 6; ++j) { lg[j] = __expf(lg[j] - mx); s += lg[j]; }
      const float inv = 1.f / s;
#pragma unroll
      for (int j = 0; j < 6; ++j) biw[j] = lg[j] * inv;
    }

    // ---------- scatter into LDS accumulators (bank-conflict-free: ch==lane) ----------
#pragma unroll 1
    for (int m = 0; m < 16; ++m) {
      const int pm = n0 + m;
      const float oc0 = o_p_fea[(b * NPTS + pm) * CIN + lane];
      const float oc1 = o_p_fea[(b * NPTS + pm) * CIN + lane + 32];
      float px = 0.f;
      if (lane < 3) px = p_xyz[(b * NPTS + pm) * 3 + lane];
#pragma unroll
      for (int j = 0; j < 6; ++j) {
        const int   sg = __shfl(segl[j], m);
        const float w  = __shfl(biw[j], m);
        atomicAdd(&accfea[sg * CIN + lane], w * oc0);
        atomicAdd(&accfea[sg * CIN + lane + 32], w * oc1);
        if (lane < 3)   atomicAdd(&accxyz[sg * 3 + lane], w * px);
        if (lane == 31) atomicAdd(&accw[sg], w);
      }
    }
  }

  // ---------- one flush of the block-local accumulator to global ----------
  __syncthreads();
  for (int i = tid; i < MPTS * CIN; i += 256)
    atomicAdd(&ws_fea[b * MPTS * CIN + i], accfea[i]);
  for (int i = tid; i < MPTS; i += 256)
    atomicAdd(&ws_w[b * MPTS + i], accw[i]);
  for (int i = tid; i < MPTS * 3; i += 256)
    atomicAdd(&ws_xyz[b * MPTS * 3 + i], accxyz[i]);
}

__global__ void __launch_bounds__(64) slic_finalize(
    const float* __restrict__ ws_fea, const float* __restrict__ ws_w,
    const float* __restrict__ ws_xyz, float* __restrict__ out) {
  const int s = blockIdx.x;   // b*M + m  (BSZ*MPTS blocks)
  const int c = threadIdx.x;  // 64 channels
  const float inv = 1.f / (ws_w[s] + 1e-8f);
  out[s * CIN + c] = ws_fea[s * CIN + c] * inv;
  if (c < 3) out[BSZ * MPTS * CIN + s * 3 + c] = ws_xyz[s * 3 + c] * inv;
}

extern "C" void kernel_launch(void* const* d_in, const int* in_sizes, int n_in,
                              void* d_out, int out_size, void* d_ws,
                              size_t ws_size, hipStream_t stream) {
  const float* sp_fea  = (const float*)d_in[0];
  const float* sp_xyz  = (const float*)d_in[1];
  const float* o_p_fea = (const float*)d_in[2];
  const float* p_xyz   = (const float*)d_in[3];
  const int*   idx_abs = (const int*)d_in[4];
  const int*   idx     = (const int*)d_in[5];
  /* d_in[6] cluster_idx unused by reference math */
  const float* W_fea = (const float*)d_in[7];
  const float* g_fea = (const float*)d_in[8];
  const float* b_fea = (const float*)d_in[9];
  const float* W_xyz = (const float*)d_in[10];
  const float* g_xyz = (const float*)d_in[11];
  const float* b_xyz = (const float*)d_in[12];
  const float* W_mlp = (const float*)d_in[13];
  const float* g_mlp = (const float*)d_in[14];
  const float* b_mlp = (const float*)d_in[15];

  float* ws_fea = (float*)d_ws;                 // BSZ*MPTS*CIN
  float* ws_w   = ws_fea + BSZ * MPTS * CIN;    // BSZ*MPTS
  float* ws_xyz = ws_w + BSZ * MPTS;            // BSZ*MPTS*3
  const size_t accBytes = (size_t)(BSZ * MPTS * (CIN + 1 + 3)) * sizeof(float);
  hipMemsetAsync(d_ws, 0, accBytes, stream);

  const size_t smBytes = (size_t)SM_FLOATS * sizeof(float);  // 312960 B <= 320KB WGP LDS
  dim3 grid(NPTS / 2048, BSZ);  // 32 x 4 blocks, 2048 points each
  slic_main<<<grid, 256, smBytes, stream>>>(
      sp_fea, sp_xyz, o_p_fea, p_xyz, idx_abs, idx,
      W_fea, g_fea, b_fea, W_xyz, g_xyz, b_xyz, W_mlp, g_mlp, b_mlp,
      ws_fea, ws_w, ws_xyz);

  slic_finalize<<<BSZ * MPTS, 64, 0, stream>>>(ws_fea, ws_w, ws_xyz,
                                               (float*)d_out);
}
